// ConvolutionLayer_33835752358262
// MI455X (gfx1250) — matrix operand order repaired and checked by reading, not compiled, not run on hardware
//
#include <hip/hip_runtime.h>
#include <hip/hip_bf16.h>

// ---------------------------------------------------------------------------
// WaveNet residual block for MI455X (gfx1250):
//   pre = dilated_causal_conv(input) + bias + Wcond @ cond      (256 x T)
//   z   = tanh(pre[:128]) * sigmoid(pre[128:])                  (128 x T)
//   out = Wout @ z + bias_out   (128 x T)
//   skip= Wskip @ z + bias_skip (256 x T)
// Strategy: bf16 WMMA 16x16x32 everywhere, f32 accum, activations staged in
// LDS time-major so B fragments are contiguous ds_load_b128 pairs.
// ---------------------------------------------------------------------------

typedef __bf16 bf16;
typedef __attribute__((ext_vector_type(8)))  bf16  v8bf;
typedef __attribute__((ext_vector_type(16))) bf16  v16bf;
typedef __attribute__((ext_vector_type(8)))  float v8f;

#define RCH   128          // residual channels
#define TWO_R 256
#define CCOND 80
#define CPAD  96           // cond channels padded to multiple of 32
#define SCH   256          // skip channels
#define TLEN  16000
#define BATCH 8
#define NT    64           // time steps per workgroup
#define HALO  8            // (K-1)*dilation = 8 left halo
#define KPRE  480          // fused K for phase 1: 3*128 conv taps + 96 cond

// d_ws layout (bf16, all 16B aligned)
#define WPRE_OFF  0                         // [256][480]
#define WSKIP_OFF (TWO_R * KPRE * 2)        // 245760: [256][128]
#define WOUT_OFF  (WSKIP_OFF + SCH*RCH*2)   // 311296: [128][128]

// ---------------- fragment helpers (contiguous k order, same for A and B) ----------

__device__ __forceinline__ v16bf frag_join(const bf16* __restrict__ p0,
                                           const bf16* __restrict__ p1) {
  v8bf lo = *(const v8bf*)p0;   // 128-bit load
  v8bf hi = *(const v8bf*)p1;   // 128-bit load
  return __builtin_shufflevector(lo, hi, 0,1,2,3,4,5,6,7,8,9,10,11,12,13,14,15);
}

// A matrix 16x32 bf16, row-major [M][ld] at (m0,k0).
// lanes 0-15: M=lane,   K = k0..k0+15 (contiguous per lane)
// lanes16-31: M=lane-16,K = k0+16..k0+31
__device__ __forceinline__ v16bf load_afrag(const bf16* __restrict__ W, int ld,
                                            int m0, int k0, int lane) {
  int m  = m0 + (lane & 15);
  int kh = (lane >> 4) * 16;
  const bf16* base = W + m * ld + k0 + kh;
  return frag_join(base, base + 8);
}

// B matrix 32x16 bf16 taken from time-major LDS tile [rows][ld(channels)].
// lanes 0-15: N=lane,   K = k0..k0+15 (contiguous per lane)
// lanes16-31: N=lane-16,K = k0+16..k0+31
__device__ __forceinline__ v16bf load_bfrag(const bf16* __restrict__ Z, int ld,
                                            int row0, int k0, int lane) {
  int n  = lane & 15;
  int kh = (lane >> 4) * 16;
  const bf16* base = Z + (row0 + n) * ld + k0 + kh;
  return frag_join(base, base + 8);
}

__device__ __forceinline__ v8f wmma_bf16(v16bf a, v16bf b, v8f c) {
  return __builtin_amdgcn_wmma_f32_16x16x32_bf16(false, a, false, b,
                                                 (short)0, c, false, false);
}

__device__ __forceinline__ float sigmoid_f(float x) {
  return 1.0f / (1.0f + __expf(-x));
}

// ---------------- weight pre-pack: fp32 -> bf16, fused layouts -------------

__global__ void wn_pack_weights(const float* __restrict__ wconv,  // [256][128][3]
                                const float* __restrict__ wcond,  // [256][80]
                                const float* __restrict__ wskip,  // [256][128]
                                const float* __restrict__ wout,   // [128][128]
                                bf16* __restrict__ Wpre,          // [256][480]
                                bf16* __restrict__ Wskip,         // [256][128]
                                bf16* __restrict__ Wout) {        // [128][128]
  int o = blockIdx.x;                       // 0..255
  for (int col = threadIdx.x; col < KPRE; col += blockDim.x) {
    float v;
    if (col < 3 * RCH) {                    // conv taps, K = [tap0|tap1|tap2]
      int j = col >> 7;                     // tap 0..2
      int i = col & 127;                    // in channel
      v = wconv[(o * RCH + i) * 3 + j];
    } else {
      int i = col - 3 * RCH;                // cond channel (padded)
      v = (i < CCOND) ? wcond[o * CCOND + i] : 0.0f;
    }
    Wpre[o * KPRE + col] = (bf16)v;
  }
  for (int col = threadIdx.x; col < RCH; col += blockDim.x) {
    Wskip[o * RCH + col] = (bf16)wskip[o * RCH + col];
    if (o < RCH) Wout[o * RCH + col] = (bf16)wout[o * RCH + col];
  }
}

// ---------------- main fused kernel ----------------------------------------

__global__ __launch_bounds__(256, 2)
void wn_block_kernel(const float* __restrict__ input,      // [8][128][16000]
                     const float* __restrict__ cond,       // [8][80][16000]
                     const float* __restrict__ bias_conv,  // [256]
                     const float* __restrict__ bias_out,   // [128]
                     const float* __restrict__ bias_skip,  // [256]
                     const bf16*  __restrict__ Wpre,       // [256][480]
                     const bf16*  __restrict__ Wskip,      // [256][128]
                     const bf16*  __restrict__ Wout,       // [128][128]
                     float* __restrict__ out,              // [8][128][16000]
                     float* __restrict__ skip) {           // [8][256][16000]
  __shared__ bf16 lds_x[(NT + HALO) * RCH];   // input tile, time-major, 18.0 KB
  __shared__ bf16 lds_c[NT * CPAD];           // cond tile,  time-major, 12.0 KB
  __shared__ bf16 lds_z[NT * RCH];            // gated act,  time-major, 16.0 KB

  const int b    = blockIdx.y;
  const int tb   = blockIdx.x * NT;           // first output time of this tile
  const int tid  = threadIdx.x;
  const int lane = tid & 31;
  const int w    = tid >> 5;                  // wave 0..7

  // ---- stage activations to LDS as bf16 (per-thread contiguous t runs) ----
  for (int idx = tid; idx < (NT + HALO) * RCH; idx += 256) {
    int c  = idx / (NT + HALO);
    int tt = idx - c * (NT + HALO);
    int gt = tb - HALO + tt;                  // global time (halo may be <0)
    float v = (gt >= 0) ? input[((size_t)b * RCH + c) * TLEN + gt] : 0.0f;
    lds_x[tt * RCH + c] = (bf16)v;
  }
  for (int idx = tid; idx < NT * CPAD; idx += 256) {
    int cc = idx / NT;
    int tt = idx - cc * NT;
    float v = (cc < CCOND) ? cond[((size_t)b * CCOND + cc) * TLEN + tb + tt] : 0.0f;
    lds_c[tt * CPAD + cc] = (bf16)v;
  }
  __syncthreads();

  // ---- phase 1: pre-activation GEMM + gate. Wave w owns gated row pair ----
  const int m0a = 16 * w;                     // tanh half rows
  const int m0b = RCH + 16 * w;               // sigmoid half rows
  const int mlane = 8 * (lane >> 4);          // D layout: M = r + 8*(lane>=16)

  float biasA[8], biasB[8];
#pragma unroll
  for (int r = 0; r < 8; ++r) {
    biasA[r] = bias_conv[m0a + r + mlane];
    biasB[r] = bias_conv[m0b + r + mlane];
  }

  for (int nt = 0; nt < 4; ++nt) {
    const int n0 = nt * 16;
    v8f accA = {0.f,0.f,0.f,0.f,0.f,0.f,0.f,0.f};
    v8f accB = {0.f,0.f,0.f,0.f,0.f,0.f,0.f,0.f};

    // dilated conv taps: output t reads input t-8+4j -> lds_x row n0+n+4j
#pragma unroll
    for (int j = 0; j < 3; ++j) {
      const int row0 = n0 + 4 * j;
#pragma unroll
      for (int kk = 0; kk < 4; ++kk) {
        v16bf bf = load_bfrag(lds_x, RCH, row0, 32 * kk, lane);
        accA = wmma_bf16(load_afrag(Wpre, KPRE, m0a, 128 * j + 32 * kk, lane), bf, accA);
        accB = wmma_bf16(load_afrag(Wpre, KPRE, m0b, 128 * j + 32 * kk, lane), bf, accB);
      }
    }
    // conditioning 1x1 (cond K block lives at column 384 of Wpre)
#pragma unroll
    for (int kk = 0; kk < 3; ++kk) {
      v16bf bf = load_bfrag(lds_c, CPAD, n0, 32 * kk, lane);
      accA = wmma_bf16(load_afrag(Wpre, KPRE, m0a, 3 * RCH + 32 * kk, lane), bf, accA);
      accB = wmma_bf16(load_afrag(Wpre, KPRE, m0b, 3 * RCH + 32 * kk, lane), bf, accB);
    }

    // gate: z = tanh(preA) * sigmoid(preB); both accs share the D layout
#pragma unroll
    for (int r = 0; r < 8; ++r) {
      float za = tanhf(accA[r] + biasA[r]);
      float zb = sigmoid_f(accB[r] + biasB[r]);
      int m = m0a + r + mlane;
      int n = n0 + (lane & 15);
      lds_z[n * RCH + m] = (bf16)(za * zb);
    }
  }
  __syncthreads();

  // ---- phase 2: skip (256x128) and out (128x128) GEMMs on z ---------------
  // 96 tile-tasks = 64 skip + 32 out; 12 per wave, all waves fully active.
  for (int tsk = w; tsk < 96; tsk += 8) {
    int mt, n0;
    const bf16*  W;
    const float* bias;
    float*       dst;
    if (tsk < 64) {
      mt = tsk >> 2;  n0 = (tsk & 3) * 16;
      W = Wskip;  bias = bias_skip;
      dst = skip + (size_t)b * SCH * TLEN;
    } else {
      int u = tsk - 64;
      mt = u >> 2;    n0 = (u & 3) * 16;
      W = Wout;   bias = bias_out;
      dst = out + (size_t)b * RCH * TLEN;
    }
    v8f acc = {0.f,0.f,0.f,0.f,0.f,0.f,0.f,0.f};
#pragma unroll
    for (int kk = 0; kk < 4; ++kk) {
      v16bf bf = load_bfrag(lds_z, RCH, n0, 32 * kk, lane);
      acc = wmma_bf16(load_afrag(W, RCH, 16 * mt, 32 * kk, lane), bf, acc);
    }
#pragma unroll
    for (int r = 0; r < 8; ++r) {
      int m = 16 * mt + r + mlane;
      int n = n0 + (lane & 15);
      dst[(size_t)m * TLEN + tb + n] = acc[r] + bias[m];
    }
  }
}

// ---------------------------------------------------------------------------

extern "C" void kernel_launch(void* const* d_in, const int* in_sizes, int n_in,
                              void* d_out, int out_size, void* d_ws, size_t ws_size,
                              hipStream_t stream) {
  const float* input      = (const float*)d_in[0];
  const float* cond       = (const float*)d_in[1];
  const float* wconv      = (const float*)d_in[2];
  const float* bias_conv  = (const float*)d_in[3];
  const float* wcond      = (const float*)d_in[4];
  const float* wout       = (const float*)d_in[5];
  const float* bias_out   = (const float*)d_in[6];
  const float* wskip      = (const float*)d_in[7];
  const float* bias_skip  = (const float*)d_in[8];

  bf16* Wpre  = (bf16*)((char*)d_ws + WPRE_OFF);
  bf16* Wskip = (bf16*)((char*)d_ws + WSKIP_OFF);
  bf16* Wout  = (bf16*)((char*)d_ws + WOUT_OFF);

  float* out_p  = (float*)d_out;                                  // [8][128][16000]
  float* skip_p = (float*)d_out + (size_t)BATCH * RCH * TLEN;     // [8][256][16000]

  wn_pack_weights<<<dim3(TWO_R), dim3(256), 0, stream>>>(
      wconv, wcond, wskip, wout, Wpre, Wskip, Wout);

  dim3 grid(TLEN / NT, BATCH);   // 250 x 8
  wn_block_kernel<<<grid, dim3(256), 0, stream>>>(
      input, cond, bias_conv, bias_out, bias_skip,
      Wpre, Wskip, Wout, out_p, skip_p);
}
